// MoD_30391188586698
// MI455X (gfx1250) — matrix operand order repaired and checked by reading, not compiled, not run on hardware
//
#include <hip/hip_runtime.h>
#include <hip/hip_bf16.h>

// ---------------------------------------------------------------------------
// MoD (mixture-of-depths) block for MI455X / gfx1250, wave32, WMMA bf16-split.
// Shapes fixed by the reference: B=4, S=4096, D=1024, F=4096, cap=512.
// GEMMs: split-bf16 (hi+lo) with v_wmma_f32_16x16x32_bf16, fp32 accum,
// double-buffered LDS with software pipelining; LDS tile rows padded to
// 80B so fragment ds_load_b128s are bank-conflict-free; staging stores are
// packed v4bf (ds_store_b64).
// Workspace layout (needs ~32.2 MB):
//   [0,      64KB)  rw        : float[B*S]      router sigmoid outputs
//   [64KB,   72KB)  idx_sorted: int[B*cap]
//   [72KB,   80KB)  w_top     : float[B*cap]
//   [128KB, +32MB)  h         : float[B*cap*F]  gelu(mod_x @ W1)
// ---------------------------------------------------------------------------

#define B_   4
#define S_   4096
#define D_   1024
#define F_   4096
#define CAP  512
#define MTOT (B_ * CAP)   // 2048
#define LDK  40           // padded LDS row stride (bf16 elems) = 80 bytes

typedef __attribute__((ext_vector_type(16))) __bf16 v16bf;
typedef __attribute__((ext_vector_type(8)))  __bf16 v8bf;
typedef __attribute__((ext_vector_type(4)))  __bf16 v4bf;
typedef __attribute__((ext_vector_type(8)))  float  v8f;

// ---------------------------------------------------------------- fill zero
__global__ void mod_fill_zero(float* __restrict__ p, long n) {
  long i = (long)blockIdx.x * blockDim.x + threadIdx.x;
  long n4 = n >> 2;
  float4* p4 = (float4*)p;
  for (long j = i; j < n4; j += (long)gridDim.x * blockDim.x)
    p4[j] = make_float4(0.f, 0.f, 0.f, 0.f);
  if (i < (n & 3)) p[(n4 << 2) + i] = 0.f;
}

// ------------------------------------------------------------------- router
// one wave (32 lanes) per token; rw = sigmoid(x . w_router)
__global__ void __launch_bounds__(256)
mod_router(const float* __restrict__ x, const float* __restrict__ wr,
           float* __restrict__ rw) {
  const int wid  = threadIdx.x >> 5;
  const int lane = threadIdx.x & 31;
  const int row  = blockIdx.x * 8 + wid;            // token in [0, B*S)
  const float4* xp = (const float4*)(x + (size_t)row * D_);
  const float4* wp = (const float4*)wr;
  float acc = 0.f;
  #pragma unroll 4
  for (int i = lane; i < D_ / 4; i += 32) {
    float4 a = xp[i], b = wp[i];
    acc += a.x * b.x + a.y * b.y + a.z * b.z + a.w * b.w;
  }
  #pragma unroll
  for (int off = 16; off > 0; off >>= 1)
    acc += __shfl_down(acc, off, 32);
  if (lane == 0) rw[row] = 1.f / (1.f + expf(-acc));
}

// -------------------------------------------------------------------- top-k
// one block per batch. Stable descending rank == jax.lax.top_k order.
// idx_sorted[c] = c-th smallest selected index; order[c] = its rank;
// w_top[c] = rw[b, order[c]]  (faithful to the reference's indexing bug).
// Also accumulates the BCE aux loss into loss_out (pre-zeroed).
__global__ void __launch_bounds__(1024)
mod_topk(const float* __restrict__ rw, int* __restrict__ idx_sorted,
         float* __restrict__ w_top, float* __restrict__ loss_out) {
  __shared__ float s_rw[S_];
  __shared__ unsigned char s_sel[S_];
  __shared__ float s_red[1024];

  const int b = blockIdx.x;
  const float* rwb = rw + (size_t)b * S_;
  for (int s = threadIdx.x; s < S_; s += 1024) s_rw[s] = rwb[s];
  __syncthreads();

  float lsum = 0.f;
  int my_rank[4];
  #pragma unroll
  for (int i = 0; i < 4; ++i) {
    const int s = threadIdx.x + i * 1024;
    const float v = s_rw[s];
    int rank = 0;
    for (int t = 0; t < S_; ++t) {
      const float u = s_rw[t];
      rank += (u > v) || (u == v && t < s);
    }
    my_rank[i] = rank;
    s_sel[s] = (rank < CAP) ? 1 : 0;
    lsum += (rank < CAP) ? fmaxf(logf(v), -100.f)
                         : fmaxf(log1pf(-v), -100.f);
  }
  __syncthreads();

  #pragma unroll
  for (int i = 0; i < 4; ++i) {
    const int s = threadIdx.x + i * 1024;
    if (my_rank[i] < CAP) {
      int c = 0;
      for (int t = 0; t < s; ++t) c += s_sel[t];
      idx_sorted[b * CAP + c] = s;
      w_top[b * CAP + c] = s_rw[my_rank[i]];   // rw at sequence pos = rank
    }
  }

  s_red[threadIdx.x] = lsum;
  __syncthreads();
  for (int st = 512; st > 0; st >>= 1) {
    if (threadIdx.x < st) s_red[threadIdx.x] += s_red[threadIdx.x + st];
    __syncthreads();
  }
  if (threadIdx.x == 0)
    atomicAdd(loss_out, -s_red[0] / (float)(B_ * S_));
}

// ------------------------------------------------------------ WMMA helpers
__device__ __forceinline__ v8f wmma3(v16bf ah, v16bf al, v16bf bh, v16bf bl,
                                     v8f c) {
  c = __builtin_amdgcn_wmma_f32_16x16x32_bf16(false, ah, false, bh, (short)0,
                                              c, false, false);
  c = __builtin_amdgcn_wmma_f32_16x16x32_bf16(false, ah, false, bl, (short)0,
                                              c, false, false);
  c = __builtin_amdgcn_wmma_f32_16x16x32_bf16(false, al, false, bh, (short)0,
                                              c, false, false);
  return c;
}

// build a 16-element fragment from two 16B LDS chunks (both 16B aligned)
__device__ __forceinline__ v16bf load_frag32(const __bf16* rowp, int o0,
                                             int o1) {
  union { v16bf v; v8bf h[2]; } u;
  u.h[0] = *(const v8bf*)(rowp + o0);
  u.h[1] = *(const v8bf*)(rowp + o1);
  return u.v;
}

// branch-free gelu (tanh approx): tanh(y) = 1 - 2/(1+exp(2y)) saturates via inf
__device__ __forceinline__ float gelu_tanh(float v) {
  const float k0 = 0.7978845608028654f;   // sqrt(2/pi)
  const float y = k0 * (v + 0.044715f * v * v * v);
  const float t = 1.f - 2.f / (1.f + __expf(2.f * y));
  return 0.5f * v * (1.f + t);
}

// ----------------------------------------------------- pipelined tile stages
struct Tile4 { float4 v[4]; };

// A tile (128x32 fp32) -> LDS hi/lo, row-major [m][k], packed b64 stores
__device__ __forceinline__ void store_a_tile(const Tile4& t, int tid,
                                             __bf16 (*sh)[LDK],
                                             __bf16 (*sl)[LDK]) {
  #pragma unroll
  for (int i = 0; i < 4; ++i) {
    const int cidx = tid + i * 256;
    const int r = cidx >> 3, c4 = (cidx & 7) * 4;
    const float* f = (const float*)&t.v[i];
    v4bf hi, lo;
    #pragma unroll
    for (int j = 0; j < 4; ++j) {
      const __bf16 hb = (__bf16)f[j];
      hi[j] = hb;
      lo[j] = (__bf16)(f[j] - (float)hb);
    }
    *(v4bf*)&sh[r][c4] = hi;
    *(v4bf*)&sl[r][c4] = lo;
  }
}

// B tile: wave g loads k-row (kk + g*4 + i), lane p covers n = n0 + 4p .. +4.
// Each thread then owns 4 consecutive k for each of its 4 n -> packed stores.
__device__ __forceinline__ void load_b_tile(const float* __restrict__ W,
                                            int ldw, int kk, int n0, int tid,
                                            Tile4& t) {
  const int p = tid & 31, g = tid >> 5;
  #pragma unroll
  for (int i = 0; i < 4; ++i)
    t.v[i] = *(const float4*)(W + (size_t)(kk + g * 4 + i) * ldw + n0 + p * 4);
}

__device__ __forceinline__ void store_b_tile(const Tile4& t, int tid,
                                             __bf16 (*sh)[LDK],
                                             __bf16 (*sl)[LDK]) {
  const int p = tid & 31, g = tid >> 5;
  const int n4 = p * 4, kb = g * 4;
  const float* f = (const float*)&t;   // f[i*4+j]: k = kb+i, n = n4+j
  #pragma unroll
  for (int j = 0; j < 4; ++j) {
    v4bf hi, lo;
    #pragma unroll
    for (int i = 0; i < 4; ++i) {
      const float v = f[i * 4 + j];
      const __bf16 hb = (__bf16)v;
      hi[i] = hb;
      lo[i] = (__bf16)(v - (float)hb);
    }
    *(v4bf*)&sh[n4 + j][kb] = hi;
    *(v4bf*)&sl[n4 + j][kb] = lo;
  }
}

__device__ __forceinline__ void compute_tile(const __bf16 (*sa_h)[LDK],
                                             const __bf16 (*sa_l)[LDK],
                                             const __bf16 (*sb_h)[LDK],
                                             const __bf16 (*sb_l)[LDK],
                                             int wm, int wn, int lm, int kh,
                                             v8f acc[4][2]) {
  // A frag lane layout (ISA 7.12.2): half kh -> K = {kh*8..+8, 16+kh*8..+8}
  v16bf afh[4], afl[4];
  #pragma unroll
  for (int mt = 0; mt < 4; ++mt) {
    afh[mt] = load_frag32(&sa_h[wm + mt * 16 + lm][0], kh * 8, 16 + kh * 8);
    afl[mt] = load_frag32(&sa_l[wm + mt * 16 + lm][0], kh * 8, 16 + kh * 8);
  }
  // B frag: lane = column n, K = {kh*8..+8, 16+kh*8..+8} (matches A)
  #pragma unroll
  for (int nt = 0; nt < 2; ++nt) {
    const v16bf bfh =
        load_frag32(&sb_h[wn + nt * 16 + lm][0], kh * 8, 16 + kh * 8);
    const v16bf bfl =
        load_frag32(&sb_l[wn + nt * 16 + lm][0], kh * 8, 16 + kh * 8);
    #pragma unroll
    for (int mt = 0; mt < 4; ++mt)
      acc[mt][nt] = wmma3(afh[mt], afl[mt], bfh, bfl, acc[mt][nt]);
  }
}

// --------------------------------------------------------------- GEMM1
// h[m,n] = gelu( sum_k x[b, idx[m], k] * W1[k, n] ),  M=2048, N=F, K=D
__global__ void __launch_bounds__(256)
mod_gemm1(const float* __restrict__ x, const float* __restrict__ W1,
          const int* __restrict__ idx_sorted, float* __restrict__ h) {
  __shared__ __align__(32) __bf16 sa_h[2][128][LDK];
  __shared__ __align__(32) __bf16 sa_l[2][128][LDK];
  __shared__ __align__(32) __bf16 sb_h[2][128][LDK];
  __shared__ __align__(32) __bf16 sb_l[2][128][LDK];
  __shared__ int rowoff[128];

  const int m0 = blockIdx.x * 128;
  const int n0 = blockIdx.y * 128;
  const int tid = threadIdx.x;

  if (tid < 128) {
    const int m = m0 + tid;
    const int b = m >> 9, c = m & (CAP - 1);
    rowoff[tid] = (b * S_ + idx_sorted[b * CAP + c]) * D_;
  }
  __syncthreads();

  const int wv = tid >> 5, lane = tid & 31;
  const int wm = (wv & 1) * 64, wn = (wv >> 1) * 32;
  const int kh = lane >> 4, lm = lane & 15;

  v8f acc[4][2];
  #pragma unroll
  for (int i = 0; i < 4; ++i)
    #pragma unroll
    for (int j = 0; j < 2; ++j)
      acc[i][j] = (v8f){0.f, 0.f, 0.f, 0.f, 0.f, 0.f, 0.f, 0.f};

  Tile4 ta, tb;
  // prologue: tile 0
  #pragma unroll
  for (int i = 0; i < 4; ++i) {
    const int cidx = tid + i * 256;
    const int r = cidx >> 3, c4 = (cidx & 7) * 4;
    ta.v[i] = *(const float4*)(x + rowoff[r] + c4);
  }
  load_b_tile(W1, F_, 0, n0, tid, tb);
  store_a_tile(ta, tid, sa_h[0], sa_l[0]);
  store_b_tile(tb, tid, sb_h[0], sb_l[0]);
  __syncthreads();

  const int NK = D_ / 32;
  for (int it = 0; it < NK; ++it) {
    const int cur = it & 1;
    const bool hasnext = (it + 1) < NK;
    if (hasnext) {
      const int kk = (it + 1) * 32;
      #pragma unroll
      for (int i = 0; i < 4; ++i) {
        const int cidx = tid + i * 256;
        const int r = cidx >> 3, c4 = (cidx & 7) * 4;
        ta.v[i] = *(const float4*)(x + rowoff[r] + kk + c4);
      }
      load_b_tile(W1, F_, kk, n0, tid, tb);
    }
    compute_tile(sa_h[cur], sa_l[cur], sb_h[cur], sb_l[cur], wm, wn, lm, kh,
                 acc);
    if (hasnext) {
      store_a_tile(ta, tid, sa_h[1 - cur], sa_l[1 - cur]);
      store_b_tile(tb, tid, sb_h[1 - cur], sb_l[1 - cur]);
    }
    __syncthreads();
  }

  // epilogue: gelu + store (C layout: VGPR r -> M = r + (lane/16)*8, N = lane%16)
  #pragma unroll
  for (int mt = 0; mt < 4; ++mt)
    #pragma unroll
    for (int nt = 0; nt < 2; ++nt)
      #pragma unroll
      for (int r = 0; r < 8; ++r) {
        const int mm = m0 + wm + mt * 16 + r + kh * 8;
        const int nn = n0 + wn + nt * 16 + lm;
        h[(size_t)mm * F_ + nn] = gelu_tanh(acc[mt][nt][r]);
      }
}

// --------------------------------------------------------------- GEMM2
// out[b, idx[m], n] = w_top[m] * sum_k h[m,k] * W2[k,n],  M=2048, N=D, K=F
__global__ void __launch_bounds__(256)
mod_gemm2(const float* __restrict__ h, const float* __restrict__ W2,
          const int* __restrict__ idx_sorted, const float* __restrict__ w_top,
          float* __restrict__ out) {
  __shared__ __align__(32) __bf16 sa_h[2][128][LDK];
  __shared__ __align__(32) __bf16 sa_l[2][128][LDK];
  __shared__ __align__(32) __bf16 sb_h[2][128][LDK];
  __shared__ __align__(32) __bf16 sb_l[2][128][LDK];
  __shared__ int dstoff[128];
  __shared__ float scl[128];

  const int m0 = blockIdx.x * 128;
  const int n0 = blockIdx.y * 128;
  const int tid = threadIdx.x;

  if (tid < 128) {
    const int m = m0 + tid;
    const int b = m >> 9, c = m & (CAP - 1);
    dstoff[tid] = (b * S_ + idx_sorted[b * CAP + c]) * D_;
    scl[tid] = w_top[b * CAP + c];
  }
  __syncthreads();

  const int wv = tid >> 5, lane = tid & 31;
  const int wm = (wv & 1) * 64, wn = (wv >> 1) * 32;
  const int kh = lane >> 4, lm = lane & 15;

  v8f acc[4][2];
  #pragma unroll
  for (int i = 0; i < 4; ++i)
    #pragma unroll
    for (int j = 0; j < 2; ++j)
      acc[i][j] = (v8f){0.f, 0.f, 0.f, 0.f, 0.f, 0.f, 0.f, 0.f};

  Tile4 ta, tb;
  #pragma unroll
  for (int i = 0; i < 4; ++i) {
    const int cidx = tid + i * 256;
    const int r = cidx >> 3, c4 = (cidx & 7) * 4;
    ta.v[i] = *(const float4*)(h + (size_t)(m0 + r) * F_ + c4);
  }
  load_b_tile(W2, D_, 0, n0, tid, tb);
  store_a_tile(ta, tid, sa_h[0], sa_l[0]);
  store_b_tile(tb, tid, sb_h[0], sb_l[0]);
  __syncthreads();

  const int NK = F_ / 32;
  for (int it = 0; it < NK; ++it) {
    const int cur = it & 1;
    const bool hasnext = (it + 1) < NK;
    if (hasnext) {
      const int kk = (it + 1) * 32;
      #pragma unroll
      for (int i = 0; i < 4; ++i) {
        const int cidx = tid + i * 256;
        const int r = cidx >> 3, c4 = (cidx & 7) * 4;
        ta.v[i] = *(const float4*)(h + (size_t)(m0 + r) * F_ + kk + c4);
      }
      load_b_tile(W2, D_, kk, n0, tid, tb);
    }
    compute_tile(sa_h[cur], sa_l[cur], sb_h[cur], sb_l[cur], wm, wn, lm, kh,
                 acc);
    if (hasnext) {
      store_a_tile(ta, tid, sa_h[1 - cur], sa_l[1 - cur]);
      store_b_tile(tb, tid, sb_h[1 - cur], sb_l[1 - cur]);
    }
    __syncthreads();
  }

  #pragma unroll
  for (int mt = 0; mt < 4; ++mt)
    #pragma unroll
    for (int nt = 0; nt < 2; ++nt)
      #pragma unroll
      for (int r = 0; r < 8; ++r) {
        const int mloc = wm + mt * 16 + r + kh * 8;
        const int nn = n0 + wn + nt * 16 + lm;
        out[(size_t)dstoff[mloc] + nn] = acc[mt][nt][r] * scl[mloc];
      }
}

// -------------------------------------------------------------------- launch
extern "C" void kernel_launch(void* const* d_in, const int* in_sizes, int n_in,
                              void* d_out, int out_size, void* d_ws,
                              size_t ws_size, hipStream_t stream) {
  (void)in_sizes; (void)n_in; (void)ws_size;
  const float* x  = (const float*)d_in[0];
  const float* wr = (const float*)d_in[1];
  const float* W1 = (const float*)d_in[2];
  const float* W2 = (const float*)d_in[3];
  float* out = (float*)d_out;

  char* ws = (char*)d_ws;
  float* rw   = (float*)(ws);                    // B*S floats   (64 KB)
  int*   idx  = (int*)(ws + (64 << 10));         // B*CAP ints   (8 KB)
  float* wtop = (float*)(ws + (72 << 10));       // B*CAP floats (8 KB)
  float* h    = (float*)(ws + (128 << 10));      // MTOT*F floats (32 MB)

  float* loss = out + (size_t)B_ * S_ * D_;

  // 1) zero output (incl. loss slot)
  mod_fill_zero<<<2048, 256, 0, stream>>>(out, (long)out_size);
  // 2) router probabilities
  mod_router<<<(B_ * S_) / 8, 256, 0, stream>>>(x, wr, rw);
  // 3) top-k select + order/w_top + BCE loss
  mod_topk<<<B_, 1024, 0, stream>>>(rw, idx, wtop, loss);
  // 4) gather + GEMM1 + gelu  (M=2048, N=4096, K=1024)
  mod_gemm1<<<dim3(MTOT / 128, F_ / 128), 256, 0, stream>>>(x, W1, idx, h);
  // 5) GEMM2 + scale + scatter (M=2048, N=1024, K=4096)
  mod_gemm2<<<dim3(MTOT / 128, D_ / 128), 256, 0, stream>>>(h, W2, idx, wtop,
                                                            out);
}
